// Attention_28862180229709
// MI455X (gfx1250) — matrix operand order repaired and checked by reading, not compiled, not run on hardware
//
#include <hip/hip_runtime.h>
#include <hip/hip_bf16.h>

// MI455X / gfx1250, wave32. All matmuls via v_wmma_f32_16x16x32_bf16.
// Data movement: double-buffered GLOBAL_LOAD_ASYNC_TO_LDS_B128 staging (ASYNCcnt),
// fragments read from LDS as ds_load_b128 batched ahead of back-to-back WMMAs.
// Pipeline: transpose weights -> LN -> QKV GEMMs -> RoPE/bilinear -> flash attn -> out GEMM.
// Workspace use: ~54 MB.

typedef __bf16 bf16;
typedef __attribute__((ext_vector_type(16))) __bf16 v16bf;
typedef __attribute__((ext_vector_type(8)))  __bf16 v8bf;
typedef __attribute__((ext_vector_type(8)))  float  v8f;

#define SEQ   2048
#define DIM   1024
#define HEADS 16
#define DHEAD 64
#define INNER 1024

__device__ __forceinline__ v8f wmma_bf16(v16bf a, v16bf b, v8f c) {
  return __builtin_amdgcn_wmma_f32_16x16x32_bf16(false, a, false, b, (short)0, c, false, false);
}

// One 16B lane-slice of an async global->LDS DMA. VDST VGPR carries the LDS byte
// address (low 32 bits of the generic shared pointer == LDS offset per aperture map).
__device__ __forceinline__ void async_ld_b128(void* lds, const void* g) {
  const unsigned int  l  = (unsigned int)(unsigned long long)lds;
  const unsigned long long ga = (unsigned long long)g;
  asm volatile("global_load_async_to_lds_b128 %0, %1, off" :: "v"(l), "v"(ga) : "memory");
}
#define WAIT_ASYNC(n) asm volatile("s_wait_asynccnt " #n ::: "memory")

// A fragment: 16x32 bf16 (MxK), row-major source with leading dim ld.
// lane m = L%16; half=L/16: elems 0-7 = K half*8+0..7, elems 8-15 = K 16+half*8+0..7.
__device__ __forceinline__ v16bf load_frag_A(const bf16* base, int ld, int lane) {
  const int m = lane & 15, half = lane >> 4;
  const bf16* p = base + m * ld + half * 8;
  v8bf lo = *(const v8bf*)(p);
  v8bf hi = *(const v8bf*)(p + 16);
  v16bf r;
#pragma unroll
  for (int i = 0; i < 8; ++i) { r[i] = lo[i]; r[i + 8] = hi[i]; }
  return r;
}

// B fragment: 32x16 bf16 (KxN) sourced from BT stored N-major (BT[n][k]), leading dim ld.
// lane n = L%16; half=L/16: elems 0-7 = K half*8+0..7, elems 8-15 = K 16+half*8+0..7.
__device__ __forceinline__ v16bf load_frag_B(const bf16* baseT, int ld, int lane) {
  const int n = lane & 15, half = lane >> 4;
  const bf16* p = baseT + n * ld + half * 8;
  v8bf lo = *(const v8bf*)(p);
  v8bf hi = *(const v8bf*)(p + 16);
  v16bf r;
#pragma unroll
  for (int i = 0; i < 8; ++i) { r[i] = lo[i]; r[i + 8] = hi[i]; }
  return r;
}

// ---------------- weight transpose + fp32 -> bf16 ----------------
__global__ __launch_bounds__(256) void transpose_bf16_kernel(
    const float* __restrict__ src, bf16* __restrict__ dst, int R, int C) {
  __shared__ float tile[32][33];
  const int c0 = blockIdx.x * 32, r0 = blockIdx.y * 32;
  const int tx = threadIdx.x & 31, ty = threadIdx.x >> 5;
#pragma unroll
  for (int i = ty; i < 32; i += 8) tile[i][tx] = src[(r0 + i) * C + c0 + tx];
  __syncthreads();
#pragma unroll
  for (int i = ty; i < 32; i += 8) dst[(c0 + i) * R + r0 + tx] = (bf16)tile[tx][i];
}

// ---------------- LayerNorm -> bf16 ----------------
__global__ __launch_bounds__(256) void ln_kernel(
    const float* __restrict__ x, const float* __restrict__ gamma, bf16* __restrict__ xn) {
  const int row = blockIdx.x, tid = threadIdx.x;
  __shared__ float red[256];
  float s = 0.f;
  for (int i = tid; i < DIM; i += 256) s += x[row * DIM + i];
  red[tid] = s; __syncthreads();
  for (int off = 128; off > 0; off >>= 1) { if (tid < off) red[tid] += red[tid + off]; __syncthreads(); }
  const float mu = red[0] * (1.f / DIM);
  __syncthreads();
  float s2 = 0.f;
  for (int i = tid; i < DIM; i += 256) { float d = x[row * DIM + i] - mu; s2 += d * d; }
  red[tid] = s2; __syncthreads();
  for (int off = 128; off > 0; off >>= 1) { if (tid < off) red[tid] += red[tid + off]; __syncthreads(); }
  const float rstd = rsqrtf(red[0] * (1.f / DIM) + 1e-5f);
  for (int i = tid; i < DIM; i += 256)
    xn[row * DIM + i] = (bf16)((x[row * DIM + i] - mu) * rstd * gamma[i]);
}

// ---------------- bf16 WMMA GEMM: C[MxN] = A[MxK] * BT[NxK]^T ----------------
// Block tile 128x64, double-buffered async LDS staging (A:128x32, B:64x32 per stage).
__global__ __launch_bounds__(256) void gemm_bf16_kernel(
    const bf16* __restrict__ A, const bf16* __restrict__ BT,
    float* __restrict__ Cf, bf16* __restrict__ Cb, int M, int N, int K) {
  const int tid = threadIdx.x, lane = tid & 31, wave = tid >> 5;
  const int row0 = blockIdx.x * 128;
  const int col0 = blockIdx.y * 64;
  const int nn = lane & 15, half = lane >> 4;

  __shared__ __align__(16) bf16 sA[2][128 * 32];
  __shared__ __align__(16) bf16 sB[2][64 * 32];

  const int arow = tid >> 2, aseg = tid & 3;  // 64 rows x 4 x 16B segments per issue

  auto issue = [&](int buf, int k0) {
    async_ld_b128(sA[buf] + arow * 32 + aseg * 8,        A + (row0 + arow) * K + k0 + aseg * 8);
    async_ld_b128(sA[buf] + (arow + 64) * 32 + aseg * 8, A + (row0 + arow + 64) * K + k0 + aseg * 8);
    async_ld_b128(sB[buf] + arow * 32 + aseg * 8,        BT + (col0 + arow) * K + k0 + aseg * 8);
  };

  issue(0, 0);
  v8f acc[4] = {};
  const int nk = K / 32;
  for (int i = 0; i < nk; ++i) {
    if (i + 1 < nk) { issue((i + 1) & 1, (i + 1) * 32); WAIT_ASYNC(3); }
    else            { WAIT_ASYNC(0); }
    __syncthreads();
    const bf16* bA = sA[i & 1] + (wave * 16) * 32;
    const bf16* bB = sB[i & 1];
    // batch all fragment loads, then back-to-back WMMAs (partial dscnt waits)
    const v16bf a  = load_frag_A(bA, 32, lane);
    const v16bf b0 = load_frag_B(bB + 0 * 16 * 32, 32, lane);
    const v16bf b1 = load_frag_B(bB + 1 * 16 * 32, 32, lane);
    const v16bf b2 = load_frag_B(bB + 2 * 16 * 32, 32, lane);
    const v16bf b3 = load_frag_B(bB + 3 * 16 * 32, 32, lane);
    acc[0] = wmma_bf16(a, b0, acc[0]);
    acc[1] = wmma_bf16(a, b1, acc[1]);
    acc[2] = wmma_bf16(a, b2, acc[2]);
    acc[3] = wmma_bf16(a, b3, acc[3]);
    __syncthreads();
  }
#pragma unroll
  for (int t = 0; t < 4; ++t) {
#pragma unroll
    for (int r = 0; r < 8; ++r) {
      const int idx = (row0 + wave * 16 + r + 8 * half) * N + col0 + 16 * t + nn;
      const float v = acc[t][r];
      if (Cf) Cf[idx] = v; else Cb[idx] = (bf16)v;
    }
  }
}

// ---------------- RoPE(q,k) + per-head bilinear(k) + v transpose ----------------
__global__ __launch_bounds__(256) void rope_kernel(
    const float* __restrict__ qf, const float* __restrict__ kvf,
    const float* __restrict__ Wb,
    bf16* __restrict__ qb, bf16* __restrict__ ktb, bf16* __restrict__ vT) {
  const int pos = blockIdx.x, tid = threadIdx.x;
  __shared__ float sQ[INNER], sK[INNER], sKR[INNER];
  for (int i = tid; i < INNER; i += 256) {
    sQ[i] = qf[pos * INNER + i];
    sK[i] = kvf[pos * (2 * INNER) + i];
  }
  __syncthreads();
  for (int i = tid; i < INNER; i += 256) {
    const int h = i >> 6, d = i & 63, j = d & 31;
    // theta_j = 10000^(-2j/64); ln(10000)/32 = 0.28782313662425575
    const float theta = __expf(-(float)j * 0.28782314f);
    const float ang = (float)pos * theta;
    const float c = cosf(ang), s = sinf(ang);
    const float rq = (d & 1) ? sQ[i - 1] : -sQ[i + 1];
    const float rk = (d & 1) ? sK[i - 1] : -sK[i + 1];
    qb[(h * SEQ + pos) * DHEAD + d] = (bf16)((sQ[i] * c + rq * s) * 0.125f); // *DHEAD^-0.5
    sKR[i] = sK[i] * c + rk * s;
    vT[(h * DHEAD + d) * SEQ + pos] = (bf16)kvf[pos * (2 * INNER) + INNER + i];
  }
  __syncthreads();
  for (int o = tid; o < INNER; o += 256) {
    const int h = o >> 6, e = o & 63;
    const float* w = Wb + h * DHEAD * DHEAD + e;  // Wb[h][d][e]
    float accv = 0.f;
#pragma unroll 8
    for (int d = 0; d < DHEAD; ++d) accv += sKR[h * DHEAD + d] * w[d * DHEAD];
    ktb[(h * SEQ + pos) * DHEAD + e] = (bf16)accv;
  }
}

// ---------------- flash attention: 1 wave = 16 query rows, 32 keys/iter ----------------
// Kt (32x64) and V (64x32) tiles staged once per block via async DMA, double-buffered.
__global__ __launch_bounds__(128) void attn_kernel(
    const bf16* __restrict__ qb, const bf16* __restrict__ ktb,
    const bf16* __restrict__ vT, bf16* __restrict__ ob) {
  const int tid = threadIdx.x, lane = tid & 31, wave = tid >> 5;
  const int h = blockIdx.y;
  const int q0 = blockIdx.x * 64 + wave * 16;
  const int nn = lane & 15, half = lane >> 4;

  __shared__ __align__(16) bf16 sKt[2][32 * 64];  // [key][d]
  __shared__ __align__(16) bf16 sV[2][64 * 32];   // [d][key]
  __shared__ __align__(16) bf16 pT[4][16 * 32];
  bf16* myP = pT[wave];

  const bf16* qbase  = qb  + (h * SEQ + q0) * DHEAD;
  const bf16* ktbase = ktb + h * SEQ * DHEAD;
  const bf16* vbase  = vT  + h * DHEAD * SEQ;

  const int krow = tid >> 3, kseg = tid & 7;  // 16 rows x 8 segs per issue (rows of 128B)
  const int vrow = tid >> 2, vseg = tid & 3;  // 32 rows x 4 segs per issue (rows of 64B)

  auto issue = [&](int buf, int kb) {
    async_ld_b128(sKt[buf] + krow * 64 + kseg * 8,        ktbase + (kb + krow) * DHEAD + kseg * 8);
    async_ld_b128(sKt[buf] + (krow + 16) * 64 + kseg * 8, ktbase + (kb + 16 + krow) * DHEAD + kseg * 8);
    async_ld_b128(sV[buf] + vrow * 32 + vseg * 8,         vbase + vrow * SEQ + kb + vseg * 8);
    async_ld_b128(sV[buf] + (vrow + 32) * 32 + vseg * 8,  vbase + (vrow + 32) * SEQ + kb + vseg * 8);
  };

  const v16bf qA0 = load_frag_A(qbase, DHEAD, lane);       // d = 0..31
  const v16bf qA1 = load_frag_A(qbase + 32, DHEAD, lane);  // d = 32..63

  v8f acc0 = {}, acc1 = {}, acc2 = {}, acc3 = {};
  float rmax[8], rsum[8];
#pragma unroll
  for (int r = 0; r < 8; ++r) { rmax[r] = -3.0e38f; rsum[r] = 0.f; }

  issue(0, 0);
  const int nit = SEQ / 32;
  for (int it = 0; it < nit; ++it) {
    if (it + 1 < nit) { issue((it + 1) & 1, (it + 1) * 32); WAIT_ASYNC(4); }
    else              { WAIT_ASYNC(0); }
    __syncthreads();
    const bf16* kt0 = sKt[it & 1];
    const bf16* vv  = sV[it & 1];
    // batch score-fragment loads, then 4 WMMAs back-to-back
    const v16bf kb0 = load_frag_B(kt0, 64, lane);
    const v16bf kb1 = load_frag_B(kt0 + 32, 64, lane);
    const v16bf kb2 = load_frag_B(kt0 + 16 * 64, 64, lane);
    const v16bf kb3 = load_frag_B(kt0 + 16 * 64 + 32, 64, lane);
    v8f S0 = {}, S1 = {};
    S0 = wmma_bf16(qA0, kb0, S0);
    S0 = wmma_bf16(qA1, kb1, S0);
    S1 = wmma_bf16(qA0, kb2, S1);
    S1 = wmma_bf16(qA1, kb3, S1);
    // online softmax: rows live in (vgpr r, lane half); reduce over the 16 lanes of a half
#pragma unroll
    for (int r = 0; r < 8; ++r) {
      const float v0 = S0[r], v1 = S1[r];
      float mx = fmaxf(v0, v1);
      mx = fmaxf(mx, __shfl_xor(mx, 1));
      mx = fmaxf(mx, __shfl_xor(mx, 2));
      mx = fmaxf(mx, __shfl_xor(mx, 4));
      mx = fmaxf(mx, __shfl_xor(mx, 8));
      const float nmax = fmaxf(rmax[r], mx);
      const float p0 = __expf(v0 - nmax), p1 = __expf(v1 - nmax);
      float ps = p0 + p1;
      ps += __shfl_xor(ps, 1);
      ps += __shfl_xor(ps, 2);
      ps += __shfl_xor(ps, 4);
      ps += __shfl_xor(ps, 8);
      const float al = __expf(rmax[r] - nmax);
      rsum[r] = rsum[r] * al + ps;
      rmax[r] = nmax;
      acc0[r] *= al; acc1[r] *= al; acc2[r] *= al; acc3[r] *= al;
      myP[(r + 8 * half) * 32 + nn]      = (bf16)p0;   // C-layout -> row-major P tile
      myP[(r + 8 * half) * 32 + 16 + nn] = (bf16)p1;
    }
    __syncthreads();
    // batch P/V fragment loads, then 4 WMMAs back-to-back
    const v16bf pA  = load_frag_A(myP, 32, lane);  // 16x32 P (K = 32 keys)
    const v16bf vb0 = load_frag_B(vv + 0 * 16 * 32, 32, lane);
    const v16bf vb1 = load_frag_B(vv + 1 * 16 * 32, 32, lane);
    const v16bf vb2 = load_frag_B(vv + 2 * 16 * 32, 32, lane);
    const v16bf vb3 = load_frag_B(vv + 3 * 16 * 32, 32, lane);
    acc0 = wmma_bf16(pA, vb0, acc0);
    acc1 = wmma_bf16(pA, vb1, acc1);
    acc2 = wmma_bf16(pA, vb2, acc2);
    acc3 = wmma_bf16(pA, vb3, acc3);
    __syncthreads();
  }
#pragma unroll
  for (int r = 0; r < 8; ++r) {
    const float inv = 1.f / rsum[r];
    const int row = q0 + r + 8 * half;
    const int base = row * INNER + h * DHEAD + nn;
    ob[base +  0] = (bf16)(acc0[r] * inv);
    ob[base + 16] = (bf16)(acc1[r] * inv);
    ob[base + 32] = (bf16)(acc2[r] * inv);
    ob[base + 48] = (bf16)(acc3[r] * inv);
  }
}

extern "C" void kernel_launch(void* const* d_in, const int* in_sizes, int n_in,
                              void* d_out, int out_size, void* d_ws, size_t ws_size,
                              hipStream_t stream) {
  (void)in_sizes; (void)n_in; (void)out_size; (void)ws_size;
  const float* x     = (const float*)d_in[0];
  const float* gamma = (const float*)d_in[1];
  const float* Wq    = (const float*)d_in[2];
  const float* Wkv   = (const float*)d_in[3];
  const float* Wb    = (const float*)d_in[4];
  const float* Wo    = (const float*)d_in[5];
  float* out = (float*)d_out;

  char* ws = (char*)d_ws;
  size_t off = 0;
  auto alloc = [&](size_t bytes) -> char* {
    char* p = ws + off; off += (bytes + 255) & ~(size_t)255; return p;
  };
  bf16* xnb  = (bf16*)alloc((size_t)SEQ * DIM * 2);
  bf16* WqT  = (bf16*)alloc((size_t)INNER * DIM * 2);
  bf16* WkvT = (bf16*)alloc((size_t)2 * INNER * DIM * 2);
  bf16* WoT  = (bf16*)alloc((size_t)DIM * INNER * 2);
  float* qf  = (float*)alloc((size_t)SEQ * INNER * 4);
  float* kvf = (float*)alloc((size_t)SEQ * 2 * INNER * 4);
  bf16* qb   = (bf16*)alloc((size_t)HEADS * SEQ * DHEAD * 2);
  bf16* ktb  = (bf16*)alloc((size_t)HEADS * SEQ * DHEAD * 2);
  bf16* vT   = (bf16*)alloc((size_t)HEADS * DHEAD * SEQ * 2);
  bf16* ob   = (bf16*)alloc((size_t)SEQ * INNER * 2);

  // weights -> bf16, transposed to [N][K]
  transpose_bf16_kernel<<<dim3(INNER / 32, DIM / 32), 256, 0, stream>>>(Wq, WqT, DIM, INNER);
  transpose_bf16_kernel<<<dim3(2 * INNER / 32, DIM / 32), 256, 0, stream>>>(Wkv, WkvT, DIM, 2 * INNER);
  transpose_bf16_kernel<<<dim3(DIM / 32, INNER / 32), 256, 0, stream>>>(Wo, WoT, INNER, DIM);
  // layernorm
  ln_kernel<<<SEQ, 256, 0, stream>>>(x, gamma, xnb);
  // projections (f32 accum out)
  gemm_bf16_kernel<<<dim3(SEQ / 128, INNER / 64), 256, 0, stream>>>(xnb, WqT, qf, nullptr, SEQ, INNER, DIM);
  gemm_bf16_kernel<<<dim3(SEQ / 128, 2 * INNER / 64), 256, 0, stream>>>(xnb, WkvT, kvf, nullptr, SEQ, 2 * INNER, DIM);
  // rope + bilinear + v transpose
  rope_kernel<<<SEQ, 256, 0, stream>>>(qf, kvf, Wb, qb, ktb, vT);
  // flash attention
  attn_kernel<<<dim3(SEQ / 64, HEADS), 128, 0, stream>>>(qb, ktb, vT, ob);
  // output projection -> fp32 d_out
  gemm_bf16_kernel<<<dim3(SEQ / 128, INNER / 64), 256, 0, stream>>>(ob, WoT, out, nullptr, SEQ, INNER, DIM);
}
